// ChunkMessagePassing_56307021250764
// MI455X (gfx1250) — matrix operand order repaired and checked by reading, not compiled, not run on hardware
//
#include <hip/hip_runtime.h>
#include <hip/hip_bf16.h>
#include <math.h>

typedef __attribute__((ext_vector_type(16))) __bf16 v16bf;
typedef __attribute__((ext_vector_type(8)))  __bf16 v8bf;
typedef __attribute__((ext_vector_type(8)))  float  v8f;

#define MTOT   8192
#define DDIM   512
#define NBATCH 4096

#if defined(__has_builtin)
#if __has_builtin(__builtin_amdgcn_tensor_load_to_lds) && __has_builtin(__builtin_amdgcn_s_wait_tensorcnt)
#define USE_TDM 1
#endif
#endif
#ifndef USE_TDM
#define USE_TDM 0
#endif

__device__ __forceinline__ void bsplit(float x, __bf16& hi, __bf16& lo) {
    hi = (__bf16)x;
    lo = (__bf16)(x - (float)hi);
}

__device__ __forceinline__ unsigned pack2(__bf16 a, __bf16 b) {
    union { __bf16 h; unsigned short u; } ua, ub;
    ua.h = a; ub.h = b;
    return (unsigned)ua.u | ((unsigned)ub.u << 16);
}

__device__ __forceinline__ float gelu_exact(float x) {
    return 0.5f * x * (1.0f + erff(x * 0.70710678118654752f));
}

// transpose-stage one float4 of W into bf16 hi/lo LDS tiles [j][k]
__device__ __forceinline__ void stage_b(__bf16 (*BH)[40], __bf16 (*BL)[40],
                                        int kr, int c4, float4 v) {
    __bf16 hh, ll;
    bsplit(v.x, hh, ll); BH[c4 + 0][kr] = hh; BL[c4 + 0][kr] = ll;
    bsplit(v.y, hh, ll); BH[c4 + 1][kr] = hh; BL[c4 + 1][kr] = ll;
    bsplit(v.z, hh, ll); BH[c4 + 2][kr] = hh; BL[c4 + 2][kr] = ll;
    bsplit(v.w, hh, ll); BH[c4 + 3][kr] = hh; BL[c4 + 3][kr] = ll;
}

#if USE_TDM
typedef unsigned __attribute__((ext_vector_type(4))) uint4v;
typedef int      __attribute__((ext_vector_type(8))) int8v;
typedef int      __attribute__((ext_vector_type(4))) int4v;

// Generic LDS pointer: low 32 bits are the LDS byte address (flat aperture).
__device__ __forceinline__ unsigned lds_off_u32(const void* p) {
    return (unsigned)(unsigned long long)(uintptr_t)p;
}

// TDM: load a 32(x) x 128(y) f32 tile (row stride 512 f32) into LDS,
// padding 4 dwords after every 32 dwords -> 144B LDS row pitch.
__device__ __forceinline__ void tdm_load_tile_a(const float* gsrc, unsigned ldsoff) {
    unsigned long long ga = (unsigned long long)(uintptr_t)gsrc;
    uint4v g0;
    g0[0] = 1u;                                   // count=1 (valid user D#)
    g0[1] = ldsoff;                               // lds_addr (bytes)
    g0[2] = (unsigned)ga;                         // global_addr[31:0]
    g0[3] = (unsigned)(ga >> 32) | (2u << 30);    // global_addr[56:32] | type=2
    int8v g1;
    g1[0] = 0x07120000;        // data_size=4B, pad_en, interval=32dw, amount=4dw
    g1[1] = (int)(512u << 16); // tensor_dim0 = 512 elements
    g1[2] = (int)(8192u << 16);// tensor_dim1 = 8192 rows
    g1[3] = (int)(32u << 16);  // tile_dim0 = 32
    g1[4] = 128;               // tile_dim1 = 128, tile_dim2 = 0
    g1[5] = 512;               // tensor_dim0_stride = 512 elements
    g1[6] = 0;
    g1[7] = 0;
    int4v z4 = {0, 0, 0, 0};
    int8v z8 = {0, 0, 0, 0, 0, 0, 0, 0};
    __builtin_amdgcn_tensor_load_to_lds(g0, g1, z4, z4, z8, 0);
}
#endif

// out[m, j] = (epilogue)( X@W + bias (+ X2@W2 for dual-K) (+ resid) )
// 3xBF16 error-compensated WMMA, f32 accumulate, double-buffered LDS.
template<bool GELU>
__global__ __launch_bounds__(256)
void gemm_kernel(const float* __restrict__ X1, const float* __restrict__ X2,
                 const float* __restrict__ W,  const float* __restrict__ bias,
                 const float* __restrict__ resid, float* __restrict__ out,
                 int Ktot)
{
    // A tile kept as f32 (TDM-landed), 36-dword pitch = conflict-free b128 reads.
    __shared__ __attribute__((aligned(16))) float  Af[2][128][36];   // 36 KB
    __shared__ __attribute__((aligned(16))) __bf16 Bh[2][64][40];    // 10 KB
    __shared__ __attribute__((aligned(16))) __bf16 Bl[2][64][40];    // 10 KB

    const int tid  = threadIdx.x;
    const int lane = tid & 31;
    const int wv   = tid >> 5;
    const int lrow = lane & 15;
    const int g    = lane >> 4;
    const int m0   = blockIdx.y * 128;
    const int j0   = blockIdx.x * 64;

    // fixed B-staging coordinates for this thread (2 float4 per K-step)
    const int br0 = tid >> 4;          // k row 0..15 (second pass: +16)
    const int bc0 = (tid & 15) << 2;   // j col

    v8f acc[4];
    #pragma unroll
    for (int t = 0; t < 4; ++t) acc[t] = (v8f){0.f,0.f,0.f,0.f,0.f,0.f,0.f,0.f};

    // ---------------- prologue: stage kt = 0 into buffer 0 ----------------
#if USE_TDM
    if (wv == 0) tdm_load_tile_a(X1 + (size_t)m0 * DDIM, lds_off_u32(&Af[0][0][0]));
#else
    #pragma unroll
    for (int q = 0; q < 4; ++q) {
        int i = tid + q * 256;
        int r = i >> 3, c4 = (i & 7) << 2;
        *(float4*)&Af[0][r][c4] =
            *(const float4*)(X1 + (size_t)(m0 + r) * DDIM + c4);
    }
#endif
    {
        float4 v0 = *(const float4*)(W + (size_t)br0        * DDIM + j0 + bc0);
        float4 v1 = *(const float4*)(W + (size_t)(16 + br0) * DDIM + j0 + bc0);
        stage_b(Bh[0], Bl[0], br0, bc0, v0);
        stage_b(Bh[0], Bl[0], 16 + br0, bc0, v1);
    }
#if USE_TDM
    if (wv == 0) __builtin_amdgcn_s_wait_tensorcnt(0);
#endif
    __syncthreads();

    int p = 0;
    for (int kt = 0; kt < Ktot; kt += 32) {
        const int  ktn       = kt + 32;
        const bool have_next = ktn < Ktot;

        // ---- issue next tile's loads (overlap with WMMA below) ----
        float4 bv0, bv1;
#if !USE_TDM
        float4 apre[4];
#endif
        if (have_next) {
            const float* Xs = X1;
            int kk0 = ktn;
            if (X2 != nullptr && ktn >= DDIM) { Xs = X2; kk0 = ktn - DDIM; }
#if USE_TDM
            if (wv == 0)
                tdm_load_tile_a(Xs + (size_t)m0 * DDIM + kk0,
                                lds_off_u32(&Af[p ^ 1][0][0]));
#else
            #pragma unroll
            for (int q = 0; q < 4; ++q) {
                int i = tid + q * 256;
                int r = i >> 3, c4 = (i & 7) << 2;
                apre[q] = *(const float4*)(Xs + (size_t)(m0 + r) * DDIM + kk0 + c4);
            }
#endif
            bv0 = *(const float4*)(W + (size_t)(ktn + br0)      * DDIM + j0 + bc0);
            bv1 = *(const float4*)(W + (size_t)(ktn + 16 + br0) * DDIM + j0 + bc0);
        }

        // ---- A fragment: f32 from LDS, hi/lo split in registers ----
        const int ar = wv * 16 + lrow;
        float4 a0 = *(const float4*)&Af[p][ar][8 * g];        // K 8g..8g+3
        float4 a1 = *(const float4*)&Af[p][ar][8 * g + 4];    // K 8g+4..8g+7
        float4 a2 = *(const float4*)&Af[p][ar][16 + 8 * g];   // K 16+8g..
        float4 a3 = *(const float4*)&Af[p][ar][20 + 8 * g];
        float af[16] = {a0.x,a0.y,a0.z,a0.w, a1.x,a1.y,a1.z,a1.w,
                        a2.x,a2.y,a2.z,a2.w, a3.x,a3.y,a3.z,a3.w};
        union VB { v16bf v; unsigned u[8]; } fah, fal;
        #pragma unroll
        for (int i = 0; i < 8; ++i) {
            __bf16 h0,l0,h1,l1;
            bsplit(af[2*i],     h0, l0);
            bsplit(af[2*i + 1], h1, l1);
            fah.u[i] = pack2(h0, h1);
            fal.u[i] = pack2(l0, l1);
        }

        // ---- 4 N-tiles x 3 compensated WMMAs ----
        #pragma unroll
        for (int nt = 0; nt < 4; ++nt) {
            const int bc = nt * 16 + lrow;
            union VB8 { v16bf v; v8bf h[2]; } fbh, fbl;
            fbh.h[0] = *(const v8bf*)&Bh[p][bc][8 * g];
            fbh.h[1] = *(const v8bf*)&Bh[p][bc][16 + 8 * g];
            fbl.h[0] = *(const v8bf*)&Bl[p][bc][8 * g];
            fbl.h[1] = *(const v8bf*)&Bl[p][bc][16 + 8 * g];
            acc[nt] = __builtin_amdgcn_wmma_f32_16x16x32_bf16(
                false, fah.v, false, fbh.v, (short)0, acc[nt], false, false);
            acc[nt] = __builtin_amdgcn_wmma_f32_16x16x32_bf16(
                false, fah.v, false, fbl.v, (short)0, acc[nt], false, false);
            acc[nt] = __builtin_amdgcn_wmma_f32_16x16x32_bf16(
                false, fal.v, false, fbh.v, (short)0, acc[nt], false, false);
        }

        // ---- publish next tile into back buffer ----
        if (have_next) {
#if !USE_TDM
            #pragma unroll
            for (int q = 0; q < 4; ++q) {
                int i = tid + q * 256;
                int r = i >> 3, c4 = (i & 7) << 2;
                *(float4*)&Af[p ^ 1][r][c4] = apre[q];
            }
#endif
            stage_b(Bh[p ^ 1], Bl[p ^ 1], br0, bc0, bv0);
            stage_b(Bh[p ^ 1], Bl[p ^ 1], 16 + br0, bc0, bv1);
#if USE_TDM
            if (wv == 0) __builtin_amdgcn_s_wait_tensorcnt(0);
#endif
            __syncthreads();
            p ^= 1;
        }
    }

    // ---- epilogue: bias / gelu / residual; C/D 16x16 f32 layout ----
    #pragma unroll
    for (int nt = 0; nt < 4; ++nt) {
        const int j  = j0 + nt * 16 + lrow;
        const float bvl = bias ? bias[j] : 0.0f;
        #pragma unroll
        for (int r = 0; r < 8; ++r) {
            const int m = m0 + wv * 16 + g * 8 + r;
            float v = acc[nt][r] + bvl;
            if (GELU) v = gelu_exact(v);
            const size_t o = (size_t)m * DDIM + j;
            if (resid) v += resid[o];
            out[o] = v;
        }
    }
}

// G[n] = (1/cnt) * sum_{o=0..min(8,n)} gelu(A[n] + Bv[n-o]); in-place G==A ok.
__global__ __launch_bounds__(256)
void window_kernel(const float* __restrict__ A, const float* __restrict__ Bv,
                   float* __restrict__ G)
{
    const int idx = blockIdx.x * 256 + threadIdx.x;
    const int m = idx >> 7;
    const int d = (idx & 127) << 2;
    const int n = m & (NBATCH - 1);
    const int omax = (n < 8) ? n : 8;
    const float4 a = *(const float4*)(A + (size_t)m * DDIM + d);
    float sx = 0.f, sy = 0.f, sz = 0.f, sw = 0.f;
    for (int o = 0; o <= omax; ++o) {
        const float4 b = *(const float4*)(Bv + (size_t)(m - o) * DDIM + d);
        sx += gelu_exact(a.x + b.x);
        sy += gelu_exact(a.y + b.y);
        sz += gelu_exact(a.z + b.z);
        sw += gelu_exact(a.w + b.w);
    }
    const float inv = 1.0f / (float)(omax + 1);
    float4 r; r.x = sx*inv; r.y = sy*inv; r.z = sz*inv; r.w = sw*inv;
    *(float4*)(G + (size_t)m * DDIM + d) = r;
}

// LayerNorm: one wave32 per row of 512.
__global__ __launch_bounds__(256)
void ln_kernel(const float* __restrict__ X, const float* __restrict__ gw,
               const float* __restrict__ bw, float* __restrict__ out)
{
    const int lane = threadIdx.x & 31;
    const int wv   = threadIdx.x >> 5;
    const int m    = blockIdx.x * 8 + wv;
    const float* row = X + (size_t)m * DDIM;
    float x[16];
    float s = 0.f, s2 = 0.f;
    #pragma unroll
    for (int i = 0; i < 16; ++i) {
        x[i] = row[i * 32 + lane];
        s  += x[i];
        s2 += x[i] * x[i];
    }
    #pragma unroll
    for (int off = 16; off > 0; off >>= 1) {
        s  += __shfl_xor(s,  off, 32);
        s2 += __shfl_xor(s2, off, 32);
    }
    const float mu  = s  * (1.0f / 512.0f);
    const float var = s2 * (1.0f / 512.0f) - mu * mu;
    const float rs  = rsqrtf(var + 1e-5f);
    float* orow = out + (size_t)m * DDIM;
    #pragma unroll
    for (int i = 0; i < 16; ++i) {
        const int c = i * 32 + lane;
        orow[c] = (x[i] - mu) * rs * gw[c] + bw[c];
    }
}

extern "C" void kernel_launch(void* const* d_in, const int* in_sizes, int n_in,
                              void* d_out, int out_size, void* d_ws, size_t ws_size,
                              hipStream_t stream)
{
    const float* h_in   = (const float*)d_in[0];
    const float* msg_w1 = (const float*)d_in[1];
    const float* msg_b1 = (const float*)d_in[2];
    const float* msg_w2 = (const float*)d_in[3];
    const float* msg_b2 = (const float*)d_in[4];
    const float* upd_w1 = (const float*)d_in[5];
    const float* upd_b1 = (const float*)d_in[6];
    const float* upd_w2 = (const float*)d_in[7];
    const float* upd_b2 = (const float*)d_in[8];
    const float* ln_g   = (const float*)d_in[9];
    const float* ln_b   = (const float*)d_in[10];
    float* out  = (float*)d_out;
    float* buf0 = (float*)d_ws;                         // 16 MB
    float* buf1 = buf0 + (size_t)MTOT * DDIM;           // 16 MB

    dim3 gg(DDIM / 64, MTOT / 128), gb(256);
    dim3 wg(MTOT * (DDIM / 4) / 256), wb(256);
    dim3 lg(MTOT / 8), lb(256);

    const float* h = h_in;
    for (int r = 0; r < 3; ++r) {
        // A = h @ Wa + msg_b1
        gemm_kernel<false><<<gg, gb, 0, stream>>>(h, nullptr, msg_w1, msg_b1,
                                                  nullptr, buf0, DDIM);
        // Bv = h @ Wb
        gemm_kernel<false><<<gg, gb, 0, stream>>>(h, nullptr,
                                                  msg_w1 + (size_t)DDIM * DDIM,
                                                  nullptr, nullptr, buf1, DDIM);
        // G = mean_o gelu(A + shift(Bv))   (in-place into buf0)
        window_kernel<<<wg, wb, 0, stream>>>(buf0, buf1, buf0);
        // agg = G @ msg_w2 + msg_b2
        gemm_kernel<false><<<gg, gb, 0, stream>>>(buf0, nullptr, msg_w2, msg_b2,
                                                  nullptr, buf1, DDIM);
        // H2 = gelu(h @ U1a + agg @ U1b + upd_b1)   (dual-K 1024)
        gemm_kernel<true><<<gg, gb, 0, stream>>>(h, buf1, upd_w1, upd_b1,
                                                 nullptr, buf0, 2 * DDIM);
        // NH = H2 @ upd_w2 + upd_b2 + h
        gemm_kernel<false><<<gg, gb, 0, stream>>>(buf0, nullptr, upd_w2, upd_b2,
                                                  h, buf1, DDIM);
        // h' = LayerNorm(NH)
        ln_kernel<<<lg, lb, 0, stream>>>(buf1, ln_g, ln_b, out);
        h = out;
    }
}
